// myCMP_76768245448884
// MI455X (gfx1250) — compile-verified, matrix operand order checked by reading, not run on hardware
//
#include <hip/hip_runtime.h>
#include <math.h>

// ---- constants from the reference ----
#define F_      256
#define G_      512
#define GAMMA_  0.5f
#define DELTA_SCALE 16.0f   // F^DELTA = 256^0.5

typedef __attribute__((ext_vector_type(2)))  float    v2f;
typedef __attribute__((ext_vector_type(4)))  float    v4f;
typedef __attribute__((ext_vector_type(8)))  float    v8f;
typedef __attribute__((ext_vector_type(16))) _Float16 v16h;

#if defined(__has_builtin)
#if __has_builtin(__builtin_amdgcn_wmma_f32_16x16x4_f32)
#define USE_F32_WMMA 1
#endif
#endif

// cheap deterministic hash -> uniform [0,1)
__device__ __forceinline__ float hash01(unsigned v) {
  v *= 2654435761u;
  v ^= v >> 16;
  v *= 2246822519u;
  return (float)(v >> 8) * (1.0f / 16777216.0f);
}

// ---------------- init: zero counters + group log accumulators ----------------
__global__ void initKernel(int* __restrict__ cnt, float* __restrict__ glog) {
  int t = threadIdx.x;            // 512 threads
  cnt[t]        = 0;              // cntP[256] ++ cntN[256]
  glog[t]       = 0.0f;           // glogP[512]
  glog[t + G_]  = 0.0f;           // glogN[512]
}

// ---------------- pass 1: per-column NaN counts (coalesced) ----------------
__global__ __launch_bounds__(F_)
void countNanKernel(const float* __restrict__ bag, int* __restrict__ cnt) {
  int f = threadIdx.x;                                  // column
  size_t base = (size_t)blockIdx.x * 256 * F_ + f;      // 256 rows per block
  int c = 0;
  #pragma unroll 8
  for (int r = 0; r < 256; ++r) {
    float v = bag[base + (size_t)r * F_];
    c += (v != v) ? 1 : 0;
  }
  atomicAdd(&cnt[f], c);
}

// ------------- prep: weight normalization + extra-mask thresholds -------------
__global__ __launch_bounds__(F_)
void prepKernel(const float* __restrict__ w,
                const int* __restrict__ cntP, const int* __restrict__ cntN,
                int NP, int NN,
                float* __restrict__ wn, float* __restrict__ pP, float* __restrict__ pN) {
  __shared__ float red[F_];
  int f = threadIdx.x;
  float tw = fmaxf(w[f], 0.0f) + 0.01f;
  red[f] = tw;
  __syncthreads();
  for (int s = F_ / 2; s > 0; s >>= 1) {
    if (f < s) red[f] += red[f + s];
    __syncthreads();
  }
  wn[f] = tw / red[0];

  int   cP = cntP[f], cN = cntN[f];
  float rP = (float)cP / (float)NP;
  float rN = (float)cN / (float)NN;
  float kP = 0.0f, kN = 0.0f;
  if (rP > rN) kN = floorf((rP - rN) * (float)NN);
  else         kP = floorf((rN - rP) * (float)NP);
  pP[f] = kP / fmaxf((float)(NP - cP), 1.0f);   // P(extra-mask | non-NaN)
  pN[f] = kN / fmaxf((float)(NN - cN), 1.0f);
}

// -------- pass 2: masked weighted sq-dist row sums via WMMA + segment-sum --------
__global__ __launch_bounds__(256)
void scoreKernel(const float* __restrict__ bag, const int* __restrict__ grp,
                 const float* __restrict__ x,  const float* __restrict__ wn,
                 const float* __restrict__ pExtra, float* __restrict__ glog,
                 unsigned salt) {
  // params packed for wide LDS reads: sXW[2f]={x}, sXW[2f+1]={w}; sPp[f]=p
  __shared__ __align__(16) float sXW[2 * F_];
  __shared__ __align__(8)  float sPp[F_];
  int t = threadIdx.x;
  sXW[2 * t]     = x[t];
  sXW[2 * t + 1] = wn[t];
  sPp[t]         = pExtra[t];
  __syncthreads();

  const int lane    = t & 31;
  const int wv      = t >> 5;                       // 8 waves / block
  const int rowBase = (blockIdx.x * 8 + wv) * 16;   // 16 rows / wave
  const int row     = rowBase + (lane & 15);
  const int g       = lane >> 4;                    // lane half selects K slice
  const float* rp   = bag + (size_t)row * F_;
  const unsigned rs = ((unsigned)row * (unsigned)F_) ^ salt;

  v8f acc = {};

#ifdef USE_F32_WMMA
  // A = 16x4 f32 tile: lane L -> row L%16, K pair 2*(L/16); B = all-ones 4x16
  v2f B; B[0] = 1.0f; B[1] = 1.0f;
  const int kOff = g * 2;
  #pragma unroll 8
  for (int kb = 0; kb < F_; kb += 4) {
    const int f0 = kb + kOff;                         // even
    v2f vv = *(const v2f*)(rp + f0);                  // global_load_b64
    v4f q  = *(const v4f*)(sXW + 2 * f0);             // x0,w0,x1,w1 (b128)
    v2f pp = *(const v2f*)(sPp + f0);                 // p0,p1       (b64)

    float v0 = vv[0], v1 = vv[1];
    float h0 = hash01(rs + (unsigned)f0);
    float h1 = hash01(rs + (unsigned)f0 + 1u);
    // branchless keep: non-NaN AND not extra-masked (bitwise &, no short-circuit)
    int k0 = (int)(v0 == v0) & (int)(h0 >= pp[0]);
    int k1 = (int)(v1 == v1) & (int)(h1 >= pp[1]);
    float d0 = v0 - q[0];
    float d1 = v1 - q[2];
    v2f A;
    A[0] = k0 ? q[1] * d0 * d0 : 0.0f;   // select discards NaN product
    A[1] = k1 ? q[3] * d1 * d1 : 0.0f;
    acc = __builtin_amdgcn_wmma_f32_16x16x4_f32(
        false, A, false, B, (short)0, acc, false, false);
  }
#else
  // Fallback: f16 16x16x32 (codegen-confirmed). 16-bit A layout:
  // lane L -> row L%16; lane half g supplies K = {8g..8g+7} U {16+8g..16+8g+7}.
  v16h B;
  #pragma unroll
  for (int i = 0; i < 16; ++i) B[i] = (_Float16)1.0f;
  for (int kb = 0; kb < F_; kb += 32) {
    v16h A;
    #pragma unroll
    for (int h = 0; h < 2; ++h) {
      #pragma unroll
      for (int j = 0; j < 8; ++j) {
        int f = kb + 16 * h + 8 * g + j;
        float v  = rp[f];
        float hv = hash01(rs + (unsigned)f);
        int   k  = (int)(v == v) & (int)(hv >= sPp[f]);
        float d  = v - sXW[2 * f];
        float e  = sXW[2 * f + 1] * d * d;
        A[h * 8 + j] = (_Float16)(k ? e : 0.0f);
      }
    }
    acc = __builtin_amdgcn_wmma_f32_16x16x32_f16(
        false, A, false, B, (short)0, acc, false, false);
  }
#endif

  // D layout: acc[j] lane 0 -> row rowBase+j ; lane 16 -> row rowBase+8+j (col 0)
  if ((lane & 15) == 0) {
    int rb = rowBase + g * 8;
    #pragma unroll
    for (int j = 0; j < 8; ++j) {
      float s     = acc[j];
      float score = 1.0f - __expf(-GAMMA_ * s / DELTA_SCALE);
      score       = fmaxf(score, 1e-30f);
      atomicAdd(&glog[grp[rb + j]], __logf(score));
    }
  }
}

// ---------------- epilogue: group terms -> scalar loss ----------------
__global__ __launch_bounds__(G_)
void finalKernel(const float* __restrict__ glogP, const float* __restrict__ glogN,
                 float* __restrict__ out) {
  __shared__ float red[G_];
  int t = threadIdx.x;                   // 512 threads, one per group
  float vp = glogP[t];
  float e  = fminf(__expf(vp), 0.9999999f);
  float termP = log1pf(-e);              // log(1 - prod(scoreP in group))
  red[t] = termP + glogN[t];             // + log(prod(scoreN in group))
  __syncthreads();
  for (int s = G_ / 2; s > 0; s >>= 1) {
    if (t < s) red[t] += red[t + s];
    __syncthreads();
  }
  if (t == 0) {
    float scale = powf((float)G_, 1.4f);
    out[0] = -(red[0] / scale);
  }
}

extern "C" void kernel_launch(void* const* d_in, const int* in_sizes, int n_in,
                              void* d_out, int out_size, void* d_ws, size_t ws_size,
                              hipStream_t stream) {
  const float* bagP = (const float*)d_in[0];
  const float* bagN = (const float*)d_in[1];
  const float* x    = (const float*)d_in[2];
  const float* w    = (const float*)d_in[3];
  const int*   gP   = (const int*)d_in[4];
  const int*   gN   = (const int*)d_in[5];
  (void)n_in; (void)out_size; (void)ws_size;

  const int NP = in_sizes[0] / F_;   // 65536
  const int NN = in_sizes[1] / F_;   // 65536

  // workspace layout
  int*   cnt   = (int*)d_ws;                 // [0..255]=cntP, [256..511]=cntN
  float* fbase = (float*)d_ws + 512;
  float* wn    = fbase;                      // 256
  float* pP    = fbase + 256;                // 256
  float* pN    = fbase + 512;                // 256
  float* glogP = fbase + 768;                // 512
  float* glogN = fbase + 768 + G_;           // 512 (contiguous after glogP)

  initKernel<<<1, 512, 0, stream>>>(cnt, glogP);

  countNanKernel<<<NP / 256, F_, 0, stream>>>(bagP, cnt);
  countNanKernel<<<NN / 256, F_, 0, stream>>>(bagN, cnt + 256);

  prepKernel<<<1, F_, 0, stream>>>(w, cnt, cnt + 256, NP, NN, wn, pP, pN);

  // 16 rows/wave, 8 waves/block -> 128 rows/block
  scoreKernel<<<NP / 128, 256, 0, stream>>>(bagP, gP, x, wn, pP, glogP, 0x9E3779B9u);
  scoreKernel<<<NN / 128, 256, 0, stream>>>(bagN, gN, x, wn, pN, glogN, 0x85EBCA6Bu);

  finalKernel<<<1, G_, 0, stream>>>(glogP, glogN, (float*)d_out);
}
